// InstantaneousPrior_20289425506544
// MI455X (gfx1250) — matrix operand order repaired and checked by reading, not compiled, not run on hardware
//
#include <hip/hip_runtime.h>
#include <hip/hip_bf16.h>

// Problem sizes (fixed by the reference)
#define B_SZ  1024
#define L_SZ  32
#define K_SZ  8
#define G_SZ  8
#define C_SZ  128
#define S_SZ  32
#define DIN   104   // 2*L + K + S

typedef __attribute__((ext_vector_type(16))) __bf16 v16bf;
typedef __attribute__((ext_vector_type(8)))  float  v8f;
typedef __attribute__((ext_vector_type(4)))  unsigned int v4u;
typedef __attribute__((ext_vector_type(8)))  int v8i_t;
typedef __attribute__((ext_vector_type(4)))  int v4i_t;

#if defined(__has_builtin)
#if __has_builtin(__builtin_amdgcn_tensor_load_to_lds) && __has_builtin(__builtin_amdgcn_s_wait_tensorcnt)
#define HAVE_TDM 1
#endif
#endif

union FragU { v16bf v; uint4 q[2]; };

// A-matrix fragment (16x32 bf16), LDS row-major with row stride 128 halves.
// Lane m=lane&15 holds row (row+m); halves 0..7 -> k = kb..kb+7, halves 8..15 -> k = kb+16..kb+23,
// kb = 8*(lane>>4)  (ISA 7.12.2, 16-bit A 16x32 layout).
__device__ __forceinline__ v16bf ldsA_frag(const __bf16* base, int row, int kOff, int lane) {
  int m  = row + (lane & 15);
  int kb = kOff + ((lane >> 4) << 3);
  FragU f;
  f.q[0] = *(const uint4*)(base + m * 128 + kb);
  f.q[1] = *(const uint4*)(base + m * 128 + kb + 16);
  return f.v;
}

// B-matrix fragment (32x16 bf16) from LDS stored [n][k] (row stride 128 halves).
// Lane half h holds K kOff+8h..kOff+8h+7 (halves 0..7) and kOff+16+8h..kOff+23+8h (halves 8..15).
__device__ __forceinline__ v16bf ldsB_frag(const __bf16* base, int col, int kOff, int lane) {
  int n  = col + (lane & 15);
  int kb = kOff + ((lane >> 4) << 3);
  FragU f;
  f.q[0] = *(const uint4*)(base + n * 128 + kb);
  f.q[1] = *(const uint4*)(base + n * 128 + kb + 16);
  return f.v;
}

#ifdef HAVE_TDM
// Issue a 1-D Tensor-Data-Mover DMA: n8 x 8-byte elements, global -> LDS.
// D# per CDNA5 ISA 8.3-8.5: group0 {count=1, lds_addr, global_addr[56:0], type=2},
// group1 {data_size=3 (8B), tensor_dim0 = tile_dim0 = n8, dims 1+ unused}.
// This toolchain exposes the 6-arg builtin (g0, g1, g2, g3, g4, cpol); groups
// 2..4 are unused for a 1-D tile (tile_dim1 == 0) and are zero-filled.
__device__ __forceinline__ void tdm_load_1d(unsigned lds_byte_off, const void* gptr, unsigned n8) {
  unsigned long long ga = (unsigned long long)gptr;
  v4u g0;
  g0[0] = 1u;                                               // count=1 (valid user D#)
  g0[1] = lds_byte_off;                                     // lds_addr (bytes)
  g0[2] = (unsigned)(ga & 0xffffffffull);                   // global_addr[31:0]
  g0[3] = (unsigned)((ga >> 32) & 0x1ffffffull) | (2u << 30); // global_addr[56:32] | type=2
  v8i_t g1;
  g1[0] = 0x00030000;                                       // wg_mask=0, data_size=3 (8B)
  g1[1] = (int)((n8 & 0xffffu) << 16);                      // tensor_dim0[15:0]
  g1[2] = (int)(((n8 >> 16) & 0xffffu) | (1u << 16));       // tensor_dim0[31:16], tensor_dim1=1
  g1[3] = (int)((n8 & 0xffffu) << 16);                      // tile_dim0 = n8 (1-D tile)
  g1[4] = 0;                                                // tile_dim1/2 = 0 (unused)
  g1[5] = (int)n8;                                          // tensor_dim0_stride[31:0]
  g1[6] = 0;
  g1[7] = 0;
  v4i_t z4 = {0, 0, 0, 0};
  v8i_t z8 = {0, 0, 0, 0, 0, 0, 0, 0};
  __builtin_amdgcn_tensor_load_to_lds(g0, g1, z4, z4, z8, 0);
}
#endif

// ---------------------------------------------------------------------------
// Kernel 0: zero the output accumulator (harness poisons d_out with 0xAA).
// ---------------------------------------------------------------------------
__global__ void zero_out_kernel(float* out, int n) {
  int i = blockIdx.x * blockDim.x + threadIdx.x;
  if (i < n) out[i] = 0.f;
}

// ---------------------------------------------------------------------------
// Kernel 1: hard gumbel-softmax assignment (psi forward value is one-hot at
// argmax_k(target_params[l,k] + gumbel(u))). One block per batch, lane=latent.
// ---------------------------------------------------------------------------
__global__ void prep_assign_kernel(const float* __restrict__ target_params,
                                   const float* __restrict__ u_gumbel,
                                   int* __restrict__ assign) {
  int b = blockIdx.x;
  int l = threadIdx.x;           // 0..31
  const float* u = u_gumbel + (b * L_SZ + l) * K_SZ;
  int   best  = 0;
  float bestv = -1e30f;
  for (int k = 0; k < K_SZ; k++) {
    float g = -__logf(-__logf(u[k] + 1e-10f) + 1e-10f);
    float v = target_params[l * K_SZ + k] + g;   // argmax invariant to /TAU
    if (v > bestv) { bestv = v; best = k; }
  }
  assign[b * L_SZ + l] = best;
}

// ---------------------------------------------------------------------------
// Kernel 2: ENCO graph sample -> latent parent masks.
// mask[b,g,i] bit j = (j != i) && (u_adj[b,g,a_j,a_i] < sig(theta)*sig(gamma)).
// ---------------------------------------------------------------------------
__global__ void prep_mask_kernel(const float* __restrict__ u_adj,
                                 const float* __restrict__ theta,
                                 const float* __restrict__ gamma,
                                 const int* __restrict__ assign,
                                 unsigned* __restrict__ maskbits) {
  int bg = blockIdx.x;           // 0 .. B*G-1
  int b  = bg >> 3;
  int i  = threadIdx.x;          // 0..31
  __shared__ int a[L_SZ];
  a[i] = assign[b * L_SZ + i];
  __syncthreads();
  int ai = a[i];
  const float* ua = u_adj + (size_t)bg * (K_SZ * K_SZ);
  unsigned m = 0u;
  #pragma unroll
  for (int j = 0; j < L_SZ; j++) {
    if (j == i) continue;
    int e = a[j] * K_SZ + ai;
    float p = (1.f / (1.f + __expf(-theta[e]))) * (1.f / (1.f + __expf(-gamma[e])));
    if (ua[e] < p) m |= (1u << j);
  }
  maskbits[bg * L_SZ + i] = m;
}

// ---------------------------------------------------------------------------
// Kernel 2b: pre-convert weights to bf16 in the TDM-friendly [L][n][128]
// layout (W1 K-padded 104->128). Done once; main kernel then DMAs raw bf16.
// ---------------------------------------------------------------------------
__global__ void prep_wbf_kernel(const float* __restrict__ W1, const float* __restrict__ W2,
                                __bf16* __restrict__ W1bf, __bf16* __restrict__ W2bf) {
  int idx = blockIdx.x * 256 + threadIdx.x;    // 0 .. L*128*128-1
  int l = idx >> 14;
  int e = idx & 16383;
  int n = e >> 7, k = e & 127;
  W1bf[idx] = (k < DIN) ? (__bf16)W1[(l * C_SZ + n) * DIN + k] : (__bf16)0.f;
  W2bf[idx] = (__bf16)W2[idx];
}

// ---------------------------------------------------------------------------
// Kernel 3: fused per-latent MLP + GroupNorm + SiLU + head + KL, WMMA bf16.
// grid = (64 M-tiles, 32 latents), block = 256 (8 waves). Each wave owns a
// 16-row x 128-col strip of the 128x128 tile via 8 f32 WMMA accumulators.
// Weights arrive in LDS via two TDM DMAs overlapped with s_inp staging.
// ---------------------------------------------------------------------------
#define SMEM_A   0
#define SMEM_W1  32768
#define SMEM_W2  65536
#define SMEM_H   98304
#define SMEM_F   131072
#define SMEM_BYTES 135232

__global__ __launch_bounds__(256)
void mlp_prior_kernel(const float* __restrict__ z_sample, const int* __restrict__ target,
                      const float* __restrict__ z_mean, const float* __restrict__ z_logstd,
                      const float* __restrict__ z_shared,
                      const __bf16* __restrict__ W1bf, const float* __restrict__ b1,
                      const float* __restrict__ gn1w, const float* __restrict__ gn1b,
                      const __bf16* __restrict__ W2bf, const float* __restrict__ b2,
                      const float* __restrict__ gn2w, const float* __restrict__ gn2b,
                      const float* __restrict__ W3, const float* __restrict__ b3,
                      const float* __restrict__ tanh_scale,
                      const int* __restrict__ assign, const unsigned* __restrict__ maskbits,
                      float* __restrict__ out) {
  extern __shared__ __align__(16) char smem[];
  __bf16* sA  = (__bf16*)(smem + SMEM_A);    // 128 x 128 bf16 activations (padded s_inp)
  __bf16* sW1 = (__bf16*)(smem + SMEM_W1);   // [n][k] bf16, k padded 104->128
  __bf16* sW2 = (__bf16*)(smem + SMEM_W2);   // [n][k] bf16
  __bf16* sH  = (__bf16*)(smem + SMEM_H);    // 128 x 128 bf16 hidden
  float*  sF  = (float*)(smem + SMEM_F);
  float* sB1 = sF;        float* sG1w = sF + 128; float* sG1b = sF + 256;
  float* sB2 = sF + 384;  float* sG2w = sF + 512; float* sG2b = sF + 640;
  float* sW3 = sF + 768;  float* sMisc = sF + 1024;  // [0..1]=b3, [2..3]=tanh_scale

  const int l        = blockIdx.y;
  const int tid      = threadIdx.x;
  const int lane     = tid & 31;
  const int wave     = tid >> 5;
  const int tileBase = blockIdx.x * 128;     // first M row (M = b*G + g)

#ifdef HAVE_TDM
  // ---- kick off weight DMAs (32 KB each) on wave 0; overlap with staging ----
  if (wave == 0) {
    unsigned ldsBase = (unsigned)__builtin_amdgcn_groupstaticsize();
    tdm_load_1d(ldsBase + SMEM_W1, W1bf + (size_t)l * 16384, 4096);  // 4096 x 8B
    tdm_load_1d(ldsBase + SMEM_W2, W2bf + (size_t)l * 16384, 4096);
  }
#else
  // ---- fallback: vectorized bf16 weight copy through VGPRs ----
  for (int i = tid; i < 2048; i += 256) {
    ((uint4*)sW1)[i] = ((const uint4*)(W1bf + (size_t)l * 16384))[i];
    ((uint4*)sW2)[i] = ((const uint4*)(W2bf + (size_t)l * 16384))[i];
  }
#endif

  // ---- stage small per-latent vectors ----
  if (tid < 128) {
    sB1[tid]  = b1[l * 128 + tid];
    sG1w[tid] = gn1w[l * 128 + tid];
    sG1b[tid] = gn1b[l * 128 + tid];
    sB2[tid]  = b2[l * 128 + tid];
    sG2w[tid] = gn2w[l * 128 + tid];
    sG2b[tid] = gn2b[l * 128 + tid];
    sW3[tid]       = W3[l * 256 + tid];        // head row 0
    sW3[128 + tid] = W3[l * 256 + 128 + tid];  // head row 1
  }
  if (tid < 2) { sMisc[tid] = b3[l * 2 + tid]; sMisc[2 + tid] = tanh_scale[l * 2 + tid]; }

  // ---- stage s_inp tile (128 rows x 104 features, padded to 128) ----
  {
    int r  = tid >> 1;
    int c0 = (tid & 1) * 64;
    int M  = tileBase + r;
    int b  = M >> 3;
    unsigned mask = maskbits[M * L_SZ + l];
    int   tgt  = target[b];
    float tval = (assign[b * L_SZ + l] == tgt) ? 1.f : 0.f;
    for (int cc = 0; cc < 64; cc++) {
      int c = c0 + cc;
      float v;
      if (c < 32)       v = ((mask >> c) & 1u) ? z_sample[b * L_SZ + c] : 0.f;  // masked parents
      else if (c < 64)  v = (float)((mask >> (c - 32)) & 1u);                    // mask
      else if (c < 72)  v = ((c - 64) == tgt) ? tval : 0.f;                      // intervention
      else if (c < 104) v = z_shared[b * S_SZ + (c - 72)];                       // shared
      else              v = 0.f;                                                 // K pad
      sA[r * 128 + c] = (__bf16)v;
    }
  }

#ifdef HAVE_TDM
  if (wave == 0) __builtin_amdgcn_s_wait_tensorcnt(0);   // weights landed in LDS
#endif
  __syncthreads();

  const v8f zero = {0.f, 0.f, 0.f, 0.f, 0.f, 0.f, 0.f, 0.f};
  v8f acc[8];
  #pragma unroll
  for (int n = 0; n < 8; n++) acc[n] = zero;

  // ---- layer 1: [128x128] x [128x128]^T via v_wmma_f32_16x16x32_bf16 ----
  #pragma unroll
  for (int kk = 0; kk < 128; kk += 32) {
    v16bf a = ldsA_frag(sA, 16 * wave, kk, lane);
    #pragma unroll
    for (int n = 0; n < 8; n++) {
      v16bf bw = ldsB_frag(sW1, 16 * n, kk, lane);
      acc[n] = __builtin_amdgcn_wmma_f32_16x16x32_bf16(false, a, false, bw,
                                                       (short)0, acc[n], false, false);
    }
  }

  // ---- epilogue 1: +bias, per-row GroupNorm(C), SiLU, store bf16 to sH ----
  {
    float vals[8][8];
    #pragma unroll
    for (int n = 0; n < 8; n++) {
      float bn = sB1[16 * n + (lane & 15)];
      #pragma unroll
      for (int r = 0; r < 8; r++) vals[n][r] = acc[n][r] + bn;
    }
    #pragma unroll
    for (int r = 0; r < 8; r++) {
      float s = 0.f, sq = 0.f;
      #pragma unroll
      for (int n = 0; n < 8; n++) { float x = vals[n][r]; s += x; sq += x * x; }
      #pragma unroll
      for (int m = 1; m < 16; m <<= 1) {        // half-wave xor reduction (wave32)
        s  += __shfl_xor(s,  m, 32);
        sq += __shfl_xor(sq, m, 32);
      }
      float mean = s * (1.f / 128.f);
      float var  = sq * (1.f / 128.f) - mean * mean;
      float rs   = rsqrtf(var + 1e-5f);
      int mrow   = 16 * wave + r + ((lane & 16) ? 8 : 0);
      #pragma unroll
      for (int n = 0; n < 8; n++) {
        int nc  = 16 * n + (lane & 15);
        float x = (vals[n][r] - mean) * rs * sG1w[nc] + sG1b[nc];
        x = x / (1.f + __expf(-x));             // SiLU
        sH[mrow * 128 + nc] = (__bf16)x;
      }
    }
  }
  __syncthreads();

  #pragma unroll
  for (int n = 0; n < 8; n++) acc[n] = zero;

  // ---- layer 2 GEMM ----
  #pragma unroll
  for (int kk = 0; kk < 128; kk += 32) {
    v16bf a = ldsA_frag(sH, 16 * wave, kk, lane);
    #pragma unroll
    for (int n = 0; n < 8; n++) {
      v16bf bw = ldsB_frag(sW2, 16 * n, kk, lane);
      acc[n] = __builtin_amdgcn_wmma_f32_16x16x32_bf16(false, a, false, bw,
                                                       (short)0, acc[n], false, false);
    }
  }

  // ---- epilogue 2: GN + SiLU + head(128->2) + StableTanh + KL, atomic out ----
  {
    float vals[8][8];
    #pragma unroll
    for (int n = 0; n < 8; n++) {
      float bn = sB2[16 * n + (lane & 15)];
      #pragma unroll
      for (int r = 0; r < 8; r++) vals[n][r] = acc[n][r] + bn;
    }
    float sc0 = __expf(sMisc[2]);
    float sc1 = __expf(sMisc[3]);
    float bb0 = sMisc[0], bb1 = sMisc[1];
    #pragma unroll
    for (int r = 0; r < 8; r++) {
      float s = 0.f, sq = 0.f;
      #pragma unroll
      for (int n = 0; n < 8; n++) { float x = vals[n][r]; s += x; sq += x * x; }
      #pragma unroll
      for (int m = 1; m < 16; m <<= 1) {
        s  += __shfl_xor(s,  m, 32);
        sq += __shfl_xor(sq, m, 32);
      }
      float mean = s * (1.f / 128.f);
      float var  = sq * (1.f / 128.f) - mean * mean;
      float rs   = rsqrtf(var + 1e-5f);
      float h0 = 0.f, h1 = 0.f;
      #pragma unroll
      for (int n = 0; n < 8; n++) {
        int nc  = 16 * n + (lane & 15);
        float x = (vals[n][r] - mean) * rs * sG2w[nc] + sG2b[nc];
        x = x / (1.f + __expf(-x));
        h0 += x * sW3[nc];
        h1 += x * sW3[128 + nc];
      }
      #pragma unroll
      for (int m = 1; m < 16; m <<= 1) {
        h0 += __shfl_xor(h0, m, 32);
        h1 += __shfl_xor(h1, m, 32);
      }
      if ((lane & 15) == 0) {                   // lane 0 -> row r, lane 16 -> row r+8
        float p0 = tanhf((h0 + bb0) / sc0) * sc0;   // prior_mean
        float p1 = tanhf((h1 + bb1) / sc1) * sc1;   // prior_logstd
        int M = tileBase + 16 * wave + r + ((lane & 16) ? 8 : 0);
        int b = M >> 3;
        float zl   = z_logstd[b * L_SZ + l];
        float zm   = z_mean[b * L_SZ + l];
        float varq = __expf(2.f * zl);
        float d    = zm - p0;
        float kld  = p1 - zl + (varq + d * d) * 0.5f * __expf(-2.f * p1) - 0.5f;
        atomicAdd(&out[b], kld * (1.f / (float)G_SZ));
      }
    }
  }
}

// ---------------------------------------------------------------------------
extern "C" void kernel_launch(void* const* d_in, const int* in_sizes, int n_in,
                              void* d_out, int out_size, void* d_ws, size_t ws_size,
                              hipStream_t stream) {
  const float* z_sample      = (const float*)d_in[0];
  const int*   target        = (const int*)  d_in[1];
  const float* z_mean        = (const float*)d_in[2];
  const float* z_logstd      = (const float*)d_in[3];
  const float* z_shared      = (const float*)d_in[4];
  const float* u_gumbel      = (const float*)d_in[5];
  const float* u_adj         = (const float*)d_in[6];
  const float* target_params = (const float*)d_in[7];
  const float* enco_theta    = (const float*)d_in[8];
  const float* enco_gamma    = (const float*)d_in[9];
  const float* W1            = (const float*)d_in[10];
  const float* b1            = (const float*)d_in[11];
  const float* gn1w          = (const float*)d_in[12];
  const float* gn1b          = (const float*)d_in[13];
  const float* W2            = (const float*)d_in[14];
  const float* b2            = (const float*)d_in[15];
  const float* gn2w          = (const float*)d_in[16];
  const float* gn2b          = (const float*)d_in[17];
  const float* W3            = (const float*)d_in[18];
  const float* b3            = (const float*)d_in[19];
  const float* tanh_scale    = (const float*)d_in[20];
  float* out = (float*)d_out;

  // workspace carve:
  // [assign: B*L int][maskbits: B*G*L u32][W1bf: L*128*128 bf16][W2bf: same]
  char* ws = (char*)d_ws;
  int*      assign   = (int*)ws;
  unsigned* maskbits = (unsigned*)(ws + (size_t)B_SZ * L_SZ * 4);
  __bf16*   W1bf     = (__bf16*)(ws + (size_t)B_SZ * L_SZ * 4 + (size_t)B_SZ * G_SZ * L_SZ * 4);
  __bf16*   W2bf     = W1bf + (size_t)L_SZ * C_SZ * 128;

  zero_out_kernel<<<(B_SZ + 255) / 256, 256, 0, stream>>>(out, B_SZ);
  prep_assign_kernel<<<B_SZ, L_SZ, 0, stream>>>(target_params, u_gumbel, assign);
  prep_mask_kernel<<<B_SZ * G_SZ, L_SZ, 0, stream>>>(u_adj, enco_theta, enco_gamma,
                                                     assign, maskbits);
  prep_wbf_kernel<<<(L_SZ * C_SZ * 128) / 256, 256, 0, stream>>>(W1, W2, W1bf, W2bf);

  dim3 grid((B_SZ * G_SZ) / 128, L_SZ);   // 64 M-tiles x 32 latents
  mlp_prior_kernel<<<grid, 256, SMEM_BYTES, stream>>>(
      z_sample, target, z_mean, z_logstd, z_shared,
      W1bf, b1, gn1w, gn1b, W2bf, b2, gn2w, gn2b, W3, b3, tanh_scale,
      assign, maskbits, out);
}
